// GATGraph_14525579395556
// MI455X (gfx1250) — compile-verified, matrix operand order checked by reading, not run on hardware
//
#include <hip/hip_runtime.h>
#include <hip/hip_bf16.h>
#include <math.h>

#define HEADS 4
#define HID 32
#define OUTC 10
#define INC 128
#define NGRAPH 128
#define NEG_SLOPE 0.2f

typedef __attribute__((ext_vector_type(16))) __bf16        bf16x16;
typedef __attribute__((ext_vector_type(8)))  float         f32x8;
typedef __attribute__((ext_vector_type(4)))  float         f32x4;
typedef __attribute__((ext_vector_type(4)))  unsigned int  u32x4;

union ABFrag { bf16x16 v; u32x4 q[2]; };

__device__ __forceinline__ __bf16 f2bf(float f) {
    unsigned u = __float_as_uint(f);
    unsigned r = u + 0x7FFFu + ((u >> 16) & 1u);   // round-to-nearest-even
    unsigned short h = (unsigned short)(r >> 16);
    return __builtin_bit_cast(__bf16, h);
}

__device__ __forceinline__ void atomicMaxF(float* addr, float val) {
    // sign-split trick; valid with -inf initialization
    if (val >= 0.f) atomicMax((int*)addr, __float_as_int(val));
    else            atomicMin((unsigned int*)addr, (unsigned int)__float_as_int(val));
}

__device__ __forceinline__ float lrelu(float v) { return v > 0.f ? v : NEG_SLOPE * v; }
__device__ __forceinline__ float elu(float v)   { return v > 0.f ? v : (expf(v) - 1.f); }

// ---------------------------------------------------------------- utilities
__global__ void fill_f32(float* __restrict__ p, float v, long long n) {
    long long i = (long long)blockIdx.x * blockDim.x + threadIdx.x;
    long long st = (long long)gridDim.x * blockDim.x;
    for (; i < n; i += st) p[i] = v;
}

__global__ void prep_w1t(const float* __restrict__ W1, __bf16* __restrict__ W1T) {
    int t = blockIdx.x * blockDim.x + threadIdx.x;          // 128*128
    if (t < INC * INC) {
        int n = t >> 7, k = t & 127;
        W1T[n * INC + k] = f2bf(W1[k * INC + n]);
    }
}

__global__ void prep_w2t(const float* __restrict__ W2, __bf16* __restrict__ W2T) {
    int t = blockIdx.x * blockDim.x + threadIdx.x;          // 16*128 (cols padded to 16)
    if (t < 16 * INC) {
        int n = t >> 7, k = t & 127;
        W2T[t] = (n < OUTC) ? f2bf(W2[k * OUTC + n]) : f2bf(0.f);
    }
}

// -------------------------------------------------- layer 1 GEMM: h1 = x@W1
// block = 128 threads (4 waves). One 16-row tile per block, each wave does two
// 16-col tiles; K=128 as 4 x wmma_f32_16x16x32_bf16.
__global__ __launch_bounds__(128) void gemm1_wmma(const float* __restrict__ x,
                                                  const __bf16* __restrict__ W1T,
                                                  float* __restrict__ h1, int N) {
    __shared__ u32x4 ldsraw[256];                 // 16x128 bf16 = 4 KB
    __bf16* ldsx = (__bf16*)ldsraw;
    const int tid = threadIdx.x;
    const int rowBase = blockIdx.x * 16;
    const long long lim = (long long)N * INC;
    for (int i = tid; i < 16 * INC; i += 128) {
        long long gi = (long long)rowBase * INC + i;
        ldsx[i] = f2bf(gi < lim ? x[gi] : 0.f);
    }
    __syncthreads();

    const int wave = tid >> 5, lane = tid & 31;
    const int lrow = lane & 15;
    const int koff = (lane < 16) ? 0 : 8;         // ISA 16-bit A layout (16x32)
    const bool fullTile = (rowBase + 16 <= N);
    for (int ct = 0; ct < 2; ++ct) {
        const int colBase = (wave * 2 + ct) * 16;
        f32x8 acc = {0.f, 0.f, 0.f, 0.f, 0.f, 0.f, 0.f, 0.f};
#pragma unroll
        for (int ks = 0; ks < 4; ++ks) {
            const int kb = ks * 32;
            ABFrag a, b;
            a.q[0] = *(const u32x4*)&ldsx[lrow * INC + kb + koff];
            a.q[1] = *(const u32x4*)&ldsx[lrow * INC + kb + koff + 16];
            const __bf16* bp = &W1T[(size_t)(colBase + lrow) * INC + kb + koff];
            b.q[0] = *(const u32x4*)bp;
            b.q[1] = *(const u32x4*)(bp + 16);
            acc = __builtin_amdgcn_wmma_f32_16x16x32_bf16(false, a.v, false, b.v,
                                                          (short)0, acc, false, false);
        }
        const int col = colBase + lrow;
        const int rbase = (lane < 16) ? 0 : 8;    // ISA 32-bit C/D layout
        float* outp = h1 + (size_t)(rowBase + rbase) * INC + col;
        if (fullTile) {                            // branch-free store clause
#pragma unroll
            for (int r = 0; r < 8; ++r) outp[(size_t)r * INC] = acc[r];
        } else {
#pragma unroll
            for (int r = 0; r < 8; ++r)
                if (rowBase + rbase + r < N) outp[(size_t)r * INC] = acc[r];
        }
    }
}

// ------------------------------------------- layer 2 GEMM: g = h2@W2 (10 cols)
__global__ __launch_bounds__(128) void gemm2_wmma(const float* __restrict__ h2,
                                                  const __bf16* __restrict__ W2T,
                                                  float* __restrict__ g, int N) {
    __shared__ u32x4 ldsraw2[1024];               // 64x128 bf16 = 16 KB
    __bf16* ldsx = (__bf16*)ldsraw2;
    const int tid = threadIdx.x;
    const int rowBase = blockIdx.x * 64;
    const long long lim = (long long)N * INC;
    for (int i = tid; i < 64 * INC; i += 128) {
        long long gi = (long long)rowBase * INC + i;
        ldsx[i] = f2bf(gi < lim ? h2[gi] : 0.f);
    }
    __syncthreads();

    const int wave = tid >> 5, lane = tid & 31;
    const int lrow = lane & 15;
    const int koff = (lane < 16) ? 0 : 8;
    const int wrow = wave * 16;
    f32x8 acc = {0.f, 0.f, 0.f, 0.f, 0.f, 0.f, 0.f, 0.f};
#pragma unroll
    for (int ks = 0; ks < 4; ++ks) {
        const int kb = ks * 32;
        ABFrag a, b;
        a.q[0] = *(const u32x4*)&ldsx[(wrow + lrow) * INC + kb + koff];
        a.q[1] = *(const u32x4*)&ldsx[(wrow + lrow) * INC + kb + koff + 16];
        const __bf16* bp = &W2T[(size_t)lrow * INC + kb + koff];
        b.q[0] = *(const u32x4*)bp;
        b.q[1] = *(const u32x4*)(bp + 16);
        acc = __builtin_amdgcn_wmma_f32_16x16x32_bf16(false, a.v, false, b.v,
                                                      (short)0, acc, false, false);
    }
    const int col = lrow;
    const int rbase = (lane < 16) ? 0 : 8;
    if (col < OUTC) {
        const int nodeBase = rowBase + wrow + rbase;
        float* outp = g + (size_t)nodeBase * OUTC + col;
        if (nodeBase + 8 <= N) {                   // fast path: no per-row guard
#pragma unroll
            for (int r = 0; r < 8; ++r) outp[(size_t)r * OUTC] = acc[r];
        } else {
#pragma unroll
            for (int r = 0; r < 8; ++r)
                if (nodeBase + r < N) outp[(size_t)r * OUTC] = acc[r];
        }
    }
}

// ------------------------------------------------------------ attention dots
__global__ void att1(const float* __restrict__ h1, const float* __restrict__ as,
                     const float* __restrict__ ad, float* __restrict__ a_s,
                     float* __restrict__ a_d, int N) {
    int t = blockIdx.x * blockDim.x + threadIdx.x;
    if (t >= N * HEADS) return;
    int node = t >> 2, h = t & 3;
    const float* row = h1 + (size_t)node * INC + h * HID;
    float s = 0.f, d = 0.f;
#pragma unroll
    for (int c = 0; c < HID; ++c) {
        float v = row[c];
        s += v * as[h * HID + c];
        d += v * ad[h * HID + c];
    }
    a_s[t] = s; a_d[t] = d;
}

__global__ void att2(const float* __restrict__ g, const float* __restrict__ as,
                     const float* __restrict__ ad, float* __restrict__ a_s,
                     float* __restrict__ a_d, int N) {
    int t = blockIdx.x * blockDim.x + threadIdx.x;
    if (t >= N) return;
    const float* row = g + (size_t)t * OUTC;
    float s = 0.f, d = 0.f;
#pragma unroll
    for (int c = 0; c < OUTC; ++c) { float v = row[c]; s += v * as[c]; d += v * ad[c]; }
    a_s[t] = s; a_d[t] = d;
}

// ---------------------------------------------------------------- edge passes
__global__ void edge_max1(const int* __restrict__ esrc, const int* __restrict__ edst,
                          int E, int Etot, const float* __restrict__ a_s,
                          const float* __restrict__ a_d, float* __restrict__ m) {
    int e = blockIdx.x * blockDim.x + threadIdx.x;
    if (e >= Etot) return;
    int s = (e < E) ? esrc[e] : (e - E);
    int d = (e < E) ? edst[e] : (e - E);
    f32x4 sv = ((const f32x4*)a_s)[s];
    f32x4 dv = ((const f32x4*)a_d)[d];
#pragma unroll
    for (int h = 0; h < HEADS; ++h)
        atomicMaxF(&m[d * HEADS + h], lrelu(sv[h] + dv[h]));
}

// one wave per edge; lanes cover channels; fold alpha normalization into the
// final per-node divide (softmax shift-invariant)
__global__ __launch_bounds__(256) void edge_acc1(const int* __restrict__ esrc,
                          const int* __restrict__ edst, int E, int Etot,
                          const float* __restrict__ a_s, const float* __restrict__ a_d,
                          const float* __restrict__ m, const float* __restrict__ h1,
                          float* __restrict__ denom, float* __restrict__ acc) {
    long long gt = (long long)blockIdx.x * blockDim.x + threadIdx.x;
    int e = (int)(gt >> 5);
    int lane = (int)(gt & 31);
    if (e >= Etot) return;
    int s = (e < E) ? esrc[e] : (e - E);
    int d = (e < E) ? edst[e] : (e - E);
    const float* hs = h1 + (size_t)s * INC;
    float* ac = acc + (size_t)d * INC;
    __builtin_prefetch(hs + lane, 0, 0);           // global_prefetch_b8: warm the gather
    f32x4 sv = ((const f32x4*)a_s)[s];             // 16B broadcast loads
    f32x4 dv = ((const f32x4*)a_d)[d];
    f32x4 mv = ((const f32x4*)m)[d];
#pragma unroll
    for (int h = 0; h < HEADS; ++h) {
        float ex = expf(lrelu(sv[h] + dv[h]) - mv[h]);
        if (lane == 0) atomicAdd(&denom[d * HEADS + h], ex);
        int c = h * HID + lane;
        atomicAdd(&ac[c], ex * hs[c]);
    }
}

__global__ void edge_max2(const int* __restrict__ esrc, const int* __restrict__ edst,
                          int E, int Etot, const float* __restrict__ a_s,
                          const float* __restrict__ a_d, float* __restrict__ m) {
    int e = blockIdx.x * blockDim.x + threadIdx.x;
    if (e >= Etot) return;
    int s = (e < E) ? esrc[e] : (e - E);
    int d = (e < E) ? edst[e] : (e - E);
    atomicMaxF(&m[d], lrelu(a_s[s] + a_d[d]));
}

__global__ void edge_acc2(const int* __restrict__ esrc, const int* __restrict__ edst,
                          int E, int Etot, const float* __restrict__ a_s,
                          const float* __restrict__ a_d, const float* __restrict__ m,
                          const float* __restrict__ g, float* __restrict__ denom,
                          float* __restrict__ acc) {
    int e = blockIdx.x * blockDim.x + threadIdx.x;
    if (e >= Etot) return;
    int s = (e < E) ? esrc[e] : (e - E);
    int d = (e < E) ? edst[e] : (e - E);
    const float* gs = g + (size_t)s * OUTC;
    __builtin_prefetch(gs, 0, 0);
    float ex = expf(lrelu(a_s[s] + a_d[d]) - m[d]);
    atomicAdd(&denom[d], ex);
    float* ac = acc + (size_t)d * OUTC;
#pragma unroll
    for (int c = 0; c < OUTC; ++c) atomicAdd(&ac[c], ex * gs[c]);
}

// ----------------------------------------------------------------- node passes
__global__ void node1(const float* __restrict__ b1, const float* __restrict__ denom,
                      float* __restrict__ acc_h2, int N) {
    long long t = (long long)blockIdx.x * blockDim.x + threadIdx.x;
    if (t >= (long long)N * INC) return;
    int node = (int)(t >> 7), c = (int)(t & 127), h = c >> 5;
    float v = acc_h2[t] / (denom[node * HEADS + h] + 1e-16f) + b1[c];
    acc_h2[t] = elu(v);
}

__global__ void node2_pool(const float* __restrict__ b2, const float* __restrict__ denom,
                           const float* __restrict__ acc, const int* __restrict__ batch,
                           float* __restrict__ pooled, float* __restrict__ cnt, int N) {
    int t = blockIdx.x * blockDim.x + threadIdx.x;
    if (t >= N) return;
    int gidx = batch[t];
    float inv = 1.f / (denom[t] + 1e-16f);
#pragma unroll
    for (int c = 0; c < OUTC; ++c) {
        float v = elu(acc[(size_t)t * OUTC + c] * inv + b2[c]);
        atomicAdd(&pooled[gidx * OUTC + c], v);
    }
    atomicAdd(&cnt[gidx], 1.f);
}

__global__ void final_k(const float* __restrict__ pooled, const float* __restrict__ cnt,
                        float* __restrict__ out) {
    int gidx = threadIdx.x;
    if (gidx >= NGRAPH) return;
    float c = fmaxf(cnt[gidx], 1.f);
    float v[OUTC];
    float mx = -__builtin_huge_valf();
#pragma unroll
    for (int i = 0; i < OUTC; ++i) { v[i] = pooled[gidx * OUTC + i] / c; mx = fmaxf(mx, v[i]); }
    float ssum = 0.f;
#pragma unroll
    for (int i = 0; i < OUTC; ++i) ssum += expf(v[i] - mx);
    float lse = mx + logf(ssum);
#pragma unroll
    for (int i = 0; i < OUTC; ++i) out[gidx * OUTC + i] = v[i] - lse;
}

// ------------------------------------------------------------------- launcher
extern "C" void kernel_launch(void* const* d_in, const int* in_sizes, int n_in,
                              void* d_out, int out_size, void* d_ws, size_t ws_size,
                              hipStream_t stream) {
    (void)n_in; (void)out_size; (void)ws_size;
    const float* x        = (const float*)d_in[0];
    const float* W1       = (const float*)d_in[1];
    const float* att_src1 = (const float*)d_in[2];
    const float* att_dst1 = (const float*)d_in[3];
    const float* b1       = (const float*)d_in[4];
    const float* W2       = (const float*)d_in[5];
    const float* att_src2 = (const float*)d_in[6];
    const float* att_dst2 = (const float*)d_in[7];
    const float* b2       = (const float*)d_in[8];
    const int*   eidx     = (const int*)d_in[9];
    const int*   batch    = (const int*)d_in[10];

    const int N    = in_sizes[10];
    const int E    = in_sizes[9] / 2;
    const int Etot = E + N;
    const int* esrc = eidx;
    const int* edst = eidx + E;

    char* ws = (char*)d_ws;
    size_t o = 0;
    auto carve = [&](size_t bytes) { size_t r = o; o = (o + bytes + 255) & ~(size_t)255; return r; };
    float*  h1     = (float*)(ws + carve((size_t)N * INC * 4));   // layer-1 features
    float*  acc1   = (float*)(ws + carve((size_t)N * INC * 4));   // msg accum -> h2 (in place)
    float*  a_s1   = (float*)(ws + carve((size_t)N * HEADS * 4));
    float*  a_d1   = (float*)(ws + carve((size_t)N * HEADS * 4));
    float*  m1     = (float*)(ws + carve((size_t)N * HEADS * 4));
    float*  den1   = (float*)(ws + carve((size_t)N * HEADS * 4));
    __bf16* W1T    = (__bf16*)(ws + carve((size_t)INC * INC * 2));
    __bf16* W2T    = (__bf16*)(ws + carve((size_t)16 * INC * 2));

    // layer-2 buffers alias the h1 region (h1 dead after edge_acc1)
    char* l2 = (char*)h1;
    size_t o2 = 0;
    auto carve2 = [&](size_t bytes) { size_t r = o2; o2 = (o2 + bytes + 255) & ~(size_t)255; return r; };
    float* g      = (float*)(l2 + carve2((size_t)N * OUTC * 4));
    float* a_s2   = (float*)(l2 + carve2((size_t)N * 4));
    float* a_d2   = (float*)(l2 + carve2((size_t)N * 4));
    float* m2     = (float*)(l2 + carve2((size_t)N * 4));
    float* den2   = (float*)(l2 + carve2((size_t)N * 4));
    float* acc2   = (float*)(l2 + carve2((size_t)N * OUTC * 4));
    float* pooled = (float*)(l2 + carve2((size_t)NGRAPH * OUTC * 4));
    float* cnt    = (float*)(l2 + carve2((size_t)NGRAPH * 4));

    const float NEG_INF = -__builtin_huge_valf();
    const int FB = 2048;  // grid-stride fill blocks

    // ---- layer 1 ----
    fill_f32<<<FB, 256, 0, stream>>>(m1, NEG_INF, (long long)N * HEADS);
    fill_f32<<<FB, 256, 0, stream>>>(den1, 0.f, (long long)N * HEADS);
    fill_f32<<<FB, 256, 0, stream>>>(acc1, 0.f, (long long)N * INC);
    prep_w1t<<<(INC * INC + 255) / 256, 256, 0, stream>>>(W1, W1T);
    prep_w2t<<<(16 * INC + 255) / 256, 256, 0, stream>>>(W2, W2T);

    gemm1_wmma<<<(N + 15) / 16, 128, 0, stream>>>(x, W1T, h1, N);
    att1<<<(N * HEADS + 255) / 256, 256, 0, stream>>>(h1, att_src1, att_dst1, a_s1, a_d1, N);
    edge_max1<<<(Etot + 255) / 256, 256, 0, stream>>>(esrc, edst, E, Etot, a_s1, a_d1, m1);
    {
        long long threads = (long long)Etot * 32;
        edge_acc1<<<(unsigned)((threads + 255) / 256), 256, 0, stream>>>(
            esrc, edst, E, Etot, a_s1, a_d1, m1, h1, den1, acc1);
    }
    node1<<<(unsigned)(((long long)N * INC + 255) / 256), 256, 0, stream>>>(b1, den1, acc1, N);

    // ---- layer 2 (aliased buffers safe after edge_acc1/node1) ----
    fill_f32<<<FB, 256, 0, stream>>>(m2, NEG_INF, N);
    fill_f32<<<FB, 256, 0, stream>>>(den2, 0.f, N);
    fill_f32<<<FB, 256, 0, stream>>>(acc2, 0.f, (long long)N * OUTC);
    fill_f32<<<8, 256, 0, stream>>>(pooled, 0.f, NGRAPH * OUTC);
    fill_f32<<<1, 256, 0, stream>>>(cnt, 0.f, NGRAPH);

    gemm2_wmma<<<(N + 63) / 64, 128, 0, stream>>>(acc1 /*h2*/, W2T, g, N);
    att2<<<(N + 255) / 256, 256, 0, stream>>>(g, att_src2, att_dst2, a_s2, a_d2, N);
    edge_max2<<<(Etot + 255) / 256, 256, 0, stream>>>(esrc, edst, E, Etot, a_s2, a_d2, m2);
    edge_acc2<<<(Etot + 255) / 256, 256, 0, stream>>>(esrc, edst, E, Etot, a_s2, a_d2, m2,
                                                      g, den2, acc2);
    node2_pool<<<(N + 255) / 256, 256, 0, stream>>>(b2, den2, acc2, batch, pooled, cnt, N);
    final_k<<<1, 128, 0, stream>>>(pooled, cnt, (float*)d_out);
}
